// DownBlock_32796370272502
// MI455X (gfx1250) — compile-verified, matrix operand order checked by reading, not run on hardware
//
#include <hip/hip_runtime.h>

// ---------------------------------------------------------------------------
// DownBlock for MI455X (gfx1250): bf16 WMMA gather-GEMM pipeline.
//   x[N,64] --BN+SiLU--> conv1(K=27,64->128) --time affine--> BN+SiLU
//          --> conv2(K=27,128->128) + x@Wid --> down conv(KD=8,128->128)
// GEMMs: v_wmma_f32_16x16x32_bf16 (16x16 tile per wave32).
// Gathers: double-buffered global_load_async_to_lds_b128 (ASYNCcnt) so the
// L2 gather latency of tap k+1 overlaps the WMMAs of tap k.
// ---------------------------------------------------------------------------

typedef __attribute__((ext_vector_type(16))) __bf16        v16bf;
typedef __attribute__((ext_vector_type(8)))  float         v8f;
typedef __attribute__((ext_vector_type(4)))  unsigned int  u32x4;

union Frag { u32x4 q[2]; v16bf v; };

static constexpr int N  = 200000;
static constexpr int NI = 64;
static constexpr int NF = 128;
static constexpr int NE = 256;
static constexpr int Bt = 16;       // batch of time embeddings
static constexpr int K  = 27;
static constexpr int KD = 8;
static constexpr int ND = N / 8;    // 25000
static constexpr float EPS = 1e-5f;
static constexpr int NCOPY = 64;    // strided BN accumulator copies

// ---- workspace layout (bytes) ---------------------------------------------
static constexpr size_t OFF_XB   = 0;                                  // bf16 x      [N,64]
static constexpr size_t OFF_H0B  = OFF_XB   + (size_t)N * NI * 2;      // bf16 bn1(x) [N,64]
static constexpr size_t OFF_BUFA = OFF_H0B  + (size_t)N * NI * 2;      // bf16 h_mid then h_final [N,128]
static constexpr size_t OFF_H2B  = OFF_BUFA + (size_t)N * NF * 2;      // bf16 bn2(h) [N,128]
static constexpr size_t OFF_W1B  = OFF_H2B  + (size_t)N * NF * 2;      // swizzled bf16 weights
static constexpr size_t OFF_W2B  = OFF_W1B  + (size_t)K  * NI * NF * 2;
static constexpr size_t OFF_WDB  = OFF_W2B  + (size_t)K  * NF * NF * 2;
static constexpr size_t OFF_WIDB = OFF_WDB  + (size_t)KD * NF * NF * 2;
static constexpr size_t OFF_TP   = OFF_WIDB + (size_t)NI * NF * 2;     // f32 tp [16,256]
static constexpr size_t OFF_ACC1 = OFF_TP   + (size_t)Bt * NE * 4;     // f32 BN1 accum (64 copies x 128)
static constexpr size_t OFF_ACC2 = OFF_ACC1 + (size_t)NCOPY * 2 * NI * 4; // f32 BN2 accum (64 copies x 256)
static constexpr size_t OFF_P1   = OFF_ACC2 + (size_t)NCOPY * 2 * NF * 4; // f32 {mean,scale} x64
static constexpr size_t OFF_P2   = OFF_P1   + (size_t)2 * NI * 4;         // f32 {mean,scale} x128
static constexpr size_t OFF_ZROW = OFF_P2   + (size_t)2 * NF * 4;         // 256B zero row for masked gathers
static constexpr size_t ZERO_SPAN_F32 = (OFF_ZROW + 256 - OFF_ACC1) / 4;  // acc1..zrow zeroed each call

// ---- helpers --------------------------------------------------------------
__device__ __forceinline__ unsigned short f2bf(float f) {
  union { float f; unsigned int u; } c; c.f = f;
  unsigned int u = c.u;
  u += 0x7fffu + ((u >> 16) & 1u);            // round-to-nearest-even
  return (unsigned short)(u >> 16);
}
__device__ __forceinline__ float bf2f(unsigned short h) {
  union { unsigned int u; float f; } c; c.u = ((unsigned int)h) << 16;
  return c.f;
}
__device__ __forceinline__ float silu(float x) { return x / (1.f + __expf(-x)); }

// Async global -> LDS 16B copy (tracked by ASYNCcnt). lds_addr = wave-relative
// LDS byte address (low 32 bits of the flat shared pointer, per ISA aperture
// truncation). Global address in a VGPR pair, GV mode (no SADDR).
__device__ __forceinline__ void async_gather16(unsigned lds_addr, const void* gptr) {
  asm volatile("global_load_async_to_lds_b128 %0, %1, off"
               :: "v"(lds_addr), "v"(gptr) : "memory");
}
__device__ __forceinline__ void wait_async0() {
  asm volatile("s_wait_asynccnt 0" ::: "memory");
}
__device__ __forceinline__ unsigned lds_off(const void* p) {
  return (unsigned)(unsigned long long)p;
}

// ---- trivial init ---------------------------------------------------------
__global__ void zero_f32(float* __restrict__ p, int n) {
  int i = blockIdx.x * blockDim.x + threadIdx.x;
  if (i < n) p[i] = 0.f;
}

// ---- weight conversion into WMMA B-fragment layout ------------------------
// Storage: frag block = ((k*nKC + kc)*nNT + nt); within block: lane*16 + j.
// B layout (16x16x32 bf16): lane<16 -> col n=lane, K(j) = {0..7,16..23};
// lane>=16 -> col n=lane-16, K(j) = {8..15,24..31}.
__global__ void cvt_weights(const float* __restrict__ W, unsigned short* __restrict__ Wb,
                            int Kk, int Ci, int Co) {
  int tid = blockIdx.x * blockDim.x + threadIdx.x;
  int E = Kk * Ci * Co;
  if (tid >= E) return;
  int j    = tid & 15;
  int lane = (tid >> 4) & 31;
  int rest = tid >> 9;
  int nNT = Co >> 4, nKC = Ci >> 5;
  int nt = rest % nNT;
  int r2 = rest / nNT;
  int kc = r2 % nKC;
  int k  = r2 / nKC;
  int hi = (lane >> 4) & 1;
  int co = nt * 16 + (lane & 15);
  int kl = (j < 8 ? j : j + 8) + (hi ? 8 : 0);
  int ci = kc * 32 + kl;
  Wb[tid] = f2bf(W[(size_t)k * Ci * Co + (size_t)ci * Co + co]);
}

// ---- BN1 statistics over x[N,64] ------------------------------------------
__global__ void __launch_bounds__(256) bn_stats_x(const float* __restrict__ x,
                                                  float* __restrict__ acc) {
  __shared__ float ls[128];
  int t = threadIdx.x;
  int c = t & 63, rg = t >> 6;
  float s = 0.f, q = 0.f;
  for (int row = blockIdx.x * 4 + rg; row < N; row += gridDim.x * 4) {
    float v = x[(size_t)row * NI + c];
    s += v; q += v * v;
  }
  if (t < 128) ls[t] = 0.f;
  __syncthreads();
  atomicAdd(&ls[c], s);
  atomicAdd(&ls[64 + c], q);
  __syncthreads();
  if (t < 128) atomicAdd(&acc[(blockIdx.x & (NCOPY - 1)) * 128 + t], ls[t]);
}

// ---- BN finalize: mean + g*rsqrt(var+eps) ---------------------------------
__global__ void bn_finalize(const float* __restrict__ acc, const float* __restrict__ g,
                            float* __restrict__ par, int C) {
  int c = threadIdx.x;
  if (c >= C) return;
  float s = 0.f, q = 0.f;
  for (int i = 0; i < NCOPY; ++i) { s += acc[i * 2 * C + c]; q += acc[i * 2 * C + C + c]; }
  float m = s / (float)N;
  float v = q / (float)N - m * m;
  par[c]     = m;
  par[C + c] = g[c] * rsqrtf(v + EPS);
}

// ---- BN1 apply + cast: h0b = bf16(silu(bn(x))), xb = bf16(x) --------------
__global__ void bn1_apply(const float* __restrict__ x, const float* __restrict__ par,
                          const float* __restrict__ be, unsigned short* __restrict__ h0b,
                          unsigned short* __restrict__ xb) {
  size_t i = (size_t)blockIdx.x * blockDim.x + threadIdx.x;
  if (i >= (size_t)N * NI) return;
  int c = (int)(i & 63);
  float v = x[i];
  h0b[i] = f2bf(silu((v - par[c]) * par[NI + c] + be[c]));
  xb[i]  = f2bf(v);
}

// ---- BN2 apply + cast: h2b = bf16(silu(bn(h_mid))) ------------------------
__global__ void bn2_apply(const unsigned short* __restrict__ hm, const float* __restrict__ par,
                          const float* __restrict__ be, unsigned short* __restrict__ h2b) {
  size_t i = (size_t)blockIdx.x * blockDim.x + threadIdx.x;
  if (i >= (size_t)N * NF) return;
  int c = (int)(i & 127);
  float v = bf2f(hm[i]);
  h2b[i] = f2bf(silu((v - par[c]) * par[NF + c] + be[c]));
}

// ---- tiny time MLP: tp = silu(t) @ Wt + bt  (16x256x256) ------------------
__global__ void __launch_bounds__(256) time_mlp(const float* __restrict__ tin,
                                                const float* __restrict__ Wt,
                                                const float* __restrict__ bt,
                                                float* __restrict__ tp) {
  __shared__ float st[Bt * NE];
  int t = threadIdx.x;
  for (int b = 0; b < Bt; ++b) st[b * NE + t] = silu(tin[b * NE + t]);
  __syncthreads();
  for (int b = 0; b < Bt; ++b) {
    float a = bt[t];
    for (int k2 = 0; k2 < NE; ++k2) a += st[b * NE + k2] * Wt[k2 * NE + t];
    tp[b * NE + t] = a;
  }
}

// ---- conv1: 27-tap gather GEMM 64->128 + time affine + BN2 partial sums ---
// Double-buffered async gather: stage tap k+1 while computing tap k.
__global__ void __launch_bounds__(256) conv1_kernel(
    const unsigned short* __restrict__ h0b, const unsigned short* __restrict__ W1b,
    const int* __restrict__ nbr, const int* __restrict__ bidx,
    const float* __restrict__ b1, const float* __restrict__ tp,
    const unsigned short* __restrict__ zrow,
    unsigned short* __restrict__ hmid, float* __restrict__ acc2) {
  __shared__ __align__(16) unsigned short a_tile[2][16 * 136]; // pad 8 -> conflict-free
  __shared__ int nshr[K * 16];
  __shared__ int bsh[16];
  const int blk = blockIdx.x, row0 = blk * 16, t = threadIdx.x;
  const int wave = t >> 5, lane = t & 31, hi = (lane >> 4) & 1, m = lane & 15;
  for (int i = t; i < K * 16; i += 256)
    nshr[i] = nbr[(i >> 4) * N + row0 + (i & 15)];
  if (t < 16) bsh[t] = bidx[row0 + t];
  __syncthreads();
  // staging role: 16 rows x 8 chunks of 8 bf16 (threads 0..127)
  const int sr = t >> 3, sc = t & 7;
  const unsigned ld0 = lds_off(&a_tile[0][sr * 136 + sc * 8]);
  const unsigned ld1 = lds_off(&a_tile[1][sr * 136 + sc * 8]);
  auto stage = [&](int k, int bufid) {
    if (t < 128) {
      int nidx = nshr[k * 16 + sr];
      const unsigned short* g = (nidx >= 0) ? (h0b + (size_t)nidx * NI + sc * 8) : zrow;
      async_gather16(bufid ? ld1 : ld0, g);
    }
  };
  v8f acc = {0.f, 0.f, 0.f, 0.f, 0.f, 0.f, 0.f, 0.f};
  stage(0, 0);
  wait_async0();
  __syncthreads();
  for (int k = 0; k < K; ++k) {
    if (k + 1 < K) stage(k + 1, (k + 1) & 1);
    const unsigned short* at = a_tile[k & 1];
#pragma unroll
    for (int kc = 0; kc < 2; ++kc) {
      Frag a, b;
      int o = m * 136 + kc * 32 + (hi ? 8 : 0);
      a.q[0] = *(const u32x4*)(at + o);
      a.q[1] = *(const u32x4*)(at + o + 16);
      const unsigned short* wp = W1b + (size_t)((k * 2 + kc) * 8 + wave) * 512 + lane * 16;
      b.q[0] = *(const u32x4*)(wp);
      b.q[1] = *(const u32x4*)(wp + 8);
      acc = __builtin_amdgcn_wmma_f32_16x16x32_bf16(false, a.v, false, b.v,
                                                    (short)0, acc, false, false);
    }
    wait_async0();
    __syncthreads();
  }
  const int col = wave * 16 + m;
  const float bias = b1[col];
  float sum = 0.f, sq = 0.f;
#pragma unroll
  for (int i = 0; i < 8; ++i) {
    int rl = i + (hi ? 8 : 0);
    int voxel = row0 + rl;
    int b = bsh[rl];
    float sc2 = tp[b * NE + col];
    float sh = tp[b * NE + NF + col];
    float v = (1.f + sc2) * (acc[i] + bias) + sh;
    hmid[(size_t)voxel * NF + col] = f2bf(v);
    sum += v; sq += v * v;
  }
  sum += __shfl_xor(sum, 16, 32);        // lane L and L^16 share a column
  sq  += __shfl_xor(sq, 16, 32);
  if (!hi) {
    float* a2 = acc2 + (blk & (NCOPY - 1)) * 2 * NF;
    atomicAdd(a2 + col, sum);
    atomicAdd(a2 + NF + col, sq);
  }
}

// ---- conv2: 27-tap gather GEMM 128->128, tap 27 = id-conv (xb @ Wid) ------
__global__ void __launch_bounds__(256) conv2_kernel(
    const unsigned short* __restrict__ h2b, const unsigned short* __restrict__ xb,
    const unsigned short* __restrict__ W2b, const unsigned short* __restrict__ Widb,
    const int* __restrict__ nbr, const float* __restrict__ b2,
    const float* __restrict__ bid, const unsigned short* __restrict__ zrow,
    unsigned short* __restrict__ hfin) {
  __shared__ __align__(16) unsigned short a_tile[2][16 * 136];
  __shared__ int nshr[(K + 1) * 16];
  const int blk = blockIdx.x, row0 = blk * 16, t = threadIdx.x;
  const int wave = t >> 5, lane = t & 31, hi = (lane >> 4) & 1, m = lane & 15;
  for (int i = t; i < (K + 1) * 16; i += 256) {
    int k = i >> 4, r = i & 15;
    nshr[i] = (k < K) ? nbr[k * N + row0 + r] : (row0 + r);   // tap K: identity
  }
  __syncthreads();
  // staging role: 16 rows x 16 chunks of 8 bf16 (all 256 threads)
  const int sr = t >> 4, sc = t & 15;
  const unsigned ld0 = lds_off(&a_tile[0][sr * 136 + sc * 8]);
  const unsigned ld1 = lds_off(&a_tile[1][sr * 136 + sc * 8]);
  auto stage = [&](int k, int bufid) {
    int nidx = nshr[k * 16 + sr];
    if (k < K) {
      const unsigned short* g = (nidx >= 0) ? (h2b + (size_t)nidx * NF + sc * 8) : zrow;
      async_gather16(bufid ? ld1 : ld0, g);
    } else if (sc < 8) {                  // identity tap: 64 channels of xb
      async_gather16(bufid ? ld1 : ld0, xb + (size_t)nidx * NI + sc * 8);
    }
  };
  v8f acc = {0.f, 0.f, 0.f, 0.f, 0.f, 0.f, 0.f, 0.f};
  stage(0, 0);
  wait_async0();
  __syncthreads();
  for (int k = 0; k <= K; ++k) {
    if (k < K) stage(k + 1, (k + 1) & 1);
    const unsigned short* at = a_tile[k & 1];
    const int nkc = (k < K) ? 4 : 2;
    const unsigned short* wbase = (k < K) ? (W2b + (size_t)k * 4 * 8 * 512) : Widb;
    for (int kc = 0; kc < nkc; ++kc) {
      Frag a, b;
      int o = m * 136 + kc * 32 + (hi ? 8 : 0);
      a.q[0] = *(const u32x4*)(at + o);
      a.q[1] = *(const u32x4*)(at + o + 16);
      const unsigned short* wp = wbase + (size_t)(kc * 8 + wave) * 512 + lane * 16;
      b.q[0] = *(const u32x4*)(wp);
      b.q[1] = *(const u32x4*)(wp + 8);
      acc = __builtin_amdgcn_wmma_f32_16x16x32_bf16(false, a.v, false, b.v,
                                                    (short)0, acc, false, false);
    }
    wait_async0();
    __syncthreads();
  }
  const int col = wave * 16 + m;
  const float bias = b2[col] + bid[col];
#pragma unroll
  for (int i = 0; i < 8; ++i) {
    int voxel = row0 + i + (hi ? 8 : 0);
    hfin[(size_t)voxel * NF + col] = f2bf(acc[i] + bias);
  }
}

// ---- strided down conv: 8-tap gather GEMM 128->128 -> f32 output ----------
__global__ void __launch_bounds__(256) conv_down_kernel(
    const unsigned short* __restrict__ hb, const unsigned short* __restrict__ Wdb,
    const int* __restrict__ nbrd, const unsigned short* __restrict__ zrow,
    float* __restrict__ out) {
  __shared__ __align__(16) unsigned short a_tile[2][16 * 136];
  __shared__ int nshr[KD * 16];
  const int blk = blockIdx.x, row0 = blk * 16, t = threadIdx.x;
  const int wave = t >> 5, lane = t & 31, hi = (lane >> 4) & 1, m = lane & 15;
  for (int i = t; i < KD * 16; i += 256) {
    int k = i >> 4, r = i & 15;
    int voxel = row0 + r;
    nshr[i] = (voxel < ND) ? nbrd[k * ND + voxel] : -1;
  }
  __syncthreads();
  const int sr = t >> 4, sc = t & 15;
  const unsigned ld0 = lds_off(&a_tile[0][sr * 136 + sc * 8]);
  const unsigned ld1 = lds_off(&a_tile[1][sr * 136 + sc * 8]);
  auto stage = [&](int k, int bufid) {
    int nidx = nshr[k * 16 + sr];
    const unsigned short* g = (nidx >= 0) ? (hb + (size_t)nidx * NF + sc * 8) : zrow;
    async_gather16(bufid ? ld1 : ld0, g);
  };
  v8f acc = {0.f, 0.f, 0.f, 0.f, 0.f, 0.f, 0.f, 0.f};
  stage(0, 0);
  wait_async0();
  __syncthreads();
  for (int k = 0; k < KD; ++k) {
    if (k + 1 < KD) stage(k + 1, (k + 1) & 1);
    const unsigned short* at = a_tile[k & 1];
#pragma unroll
    for (int kc = 0; kc < 4; ++kc) {
      Frag a, b;
      int o = m * 136 + kc * 32 + (hi ? 8 : 0);
      a.q[0] = *(const u32x4*)(at + o);
      a.q[1] = *(const u32x4*)(at + o + 16);
      const unsigned short* wp = Wdb + (size_t)((k * 4 + kc) * 8 + wave) * 512 + lane * 16;
      b.q[0] = *(const u32x4*)(wp);
      b.q[1] = *(const u32x4*)(wp + 8);
      acc = __builtin_amdgcn_wmma_f32_16x16x32_bf16(false, a.v, false, b.v,
                                                    (short)0, acc, false, false);
    }
    wait_async0();
    __syncthreads();
  }
  const int col = wave * 16 + m;
#pragma unroll
  for (int i = 0; i < 8; ++i) {
    int voxel = row0 + i + (hi ? 8 : 0);
    if (voxel < ND) out[(size_t)voxel * NF + col] = acc[i];
  }
}

// ---------------------------------------------------------------------------
extern "C" void kernel_launch(void* const* d_in, const int* in_sizes, int n_in,
                              void* d_out, int out_size, void* d_ws, size_t ws_size,
                              hipStream_t stream) {
  const float* x    = (const float*)d_in[0];
  const float* tin  = (const float*)d_in[1];
  const int*   bidx = (const int*)d_in[2];
  const int*   nbr  = (const int*)d_in[3];
  const int*   nbrd = (const int*)d_in[4];
  const float* g1   = (const float*)d_in[5];
  const float* be1  = (const float*)d_in[6];
  const float* W1   = (const float*)d_in[7];
  const float* b1   = (const float*)d_in[8];
  const float* Wt   = (const float*)d_in[9];
  const float* bt   = (const float*)d_in[10];
  const float* g2   = (const float*)d_in[11];
  const float* be2  = (const float*)d_in[12];
  const float* W2   = (const float*)d_in[13];
  const float* b2   = (const float*)d_in[14];
  const float* Wid  = (const float*)d_in[15];
  const float* bid  = (const float*)d_in[16];
  const float* Wd   = (const float*)d_in[17];
  float* out = (float*)d_out;

  char* ws = (char*)d_ws;
  unsigned short* xb   = (unsigned short*)(ws + OFF_XB);
  unsigned short* h0b  = (unsigned short*)(ws + OFF_H0B);
  unsigned short* bufA = (unsigned short*)(ws + OFF_BUFA);  // h_mid, then h_final
  unsigned short* h2b  = (unsigned short*)(ws + OFF_H2B);
  unsigned short* W1b  = (unsigned short*)(ws + OFF_W1B);
  unsigned short* W2b  = (unsigned short*)(ws + OFF_W2B);
  unsigned short* Wdb  = (unsigned short*)(ws + OFF_WDB);
  unsigned short* Widb = (unsigned short*)(ws + OFF_WIDB);
  float* tp   = (float*)(ws + OFF_TP);
  float* acc1 = (float*)(ws + OFF_ACC1);
  float* acc2 = (float*)(ws + OFF_ACC2);
  float* par1 = (float*)(ws + OFF_P1);
  float* par2 = (float*)(ws + OFF_P2);
  unsigned short* zrow = (unsigned short*)(ws + OFF_ZROW);

  // 0. zero BN accumulators + zero gather row (contiguous span)
  zero_f32<<<((int)ZERO_SPAN_F32 + 255) / 256, 256, 0, stream>>>(acc1, (int)ZERO_SPAN_F32);
  // 1. weight conversion into WMMA B-fragment layout
  cvt_weights<<<(K * NI * NF + 255) / 256, 256, 0, stream>>>(W1, W1b, K, NI, NF);
  cvt_weights<<<(K * NF * NF + 255) / 256, 256, 0, stream>>>(W2, W2b, K, NF, NF);
  cvt_weights<<<(KD * NF * NF + 255) / 256, 256, 0, stream>>>(Wd, Wdb, KD, NF, NF);
  cvt_weights<<<(NI * NF + 255) / 256, 256, 0, stream>>>(Wid, Widb, 1, NI, NF);
  // 2. BN1 stats + finalize, time MLP
  bn_stats_x<<<512, 256, 0, stream>>>(x, acc1);
  bn_finalize<<<1, NI, 0, stream>>>(acc1, g1, par1, NI);
  time_mlp<<<1, 256, 0, stream>>>(tin, Wt, bt, tp);
  // 3. BN1 apply + bf16 casts
  bn1_apply<<<(N * NI + 255) / 256, 256, 0, stream>>>(x, par1, be1, h0b, xb);
  // 4. conv1 (+ time affine, + BN2 partial sums) -> bufA (= h_mid)
  conv1_kernel<<<N / 16, 256, 0, stream>>>(h0b, W1b, nbr, bidx, b1, tp, zrow, bufA, acc2);
  // 5. BN2 finalize + apply
  bn_finalize<<<1, NF, 0, stream>>>(acc2, g2, par2, NF);
  bn2_apply<<<(N * NF + 255) / 256, 256, 0, stream>>>(bufA, par2, be2, h2b);
  // 6. conv2 + id-conv residual -> bufA (= h_final; h_mid is dead)
  conv2_kernel<<<N / 16, 256, 0, stream>>>(h2b, xb, W2b, Widb, nbr, b2, bid, zrow, bufA);
  // 7. strided down conv -> d_out
  conv_down_kernel<<<(ND + 15) / 16, 256, 0, stream>>>(bufA, Wdb, nbrd, zrow, out);
}
